// mLSTM_18124761989425
// MI455X (gfx1250) — matrix operand order repaired and checked by reading, not compiled, not run on hardware
//
#include <hip/hip_runtime.h>
#include <hip/hip_bf16.h>

// mLSTM step for MI455X (gfx1250): WMMA bf16 GEMMs (16x64 per wave, 4 WMMA/K-step)
// + fused bandwidth-bound cell-state update with non-temporal streaming.
// B=256, D=1024, H=8, HD=128, P=2048, HID=1024, K=4.

#define BB    256
#define DD    1024
#define HH    8
#define HDIM  128
#define PP    2048
#define HIDD  1024
#define KCONV 4
#define EPSV  1e-5f

typedef __bf16 bf16_t;
typedef __attribute__((ext_vector_type(16))) __bf16 v16bf;
typedef __attribute__((ext_vector_type(8)))  float  v8f;
typedef __attribute__((ext_vector_type(4)))  float  v4f;

__device__ __forceinline__ float sigmoidf_(float x) {
  return __builtin_amdgcn_rcpf(1.0f + __expf(-x));   // v_rcp_f32, no IEEE div expansion
}
__device__ __forceinline__ float siluf_(float x) { return x * sigmoidf_(x); }

// ---------------------------------------------------------------- LayerNorm
__global__ void ln_kernel(const float* __restrict__ seq, const float* __restrict__ w,
                          const float* __restrict__ b, float* __restrict__ xn) {
  __shared__ float red[256];
  const int row = blockIdx.x;
  const int tid = threadIdx.x;
  const float* x = seq + (size_t)row * DD;
  float*       y = xn  + (size_t)row * DD;
  float lx[4];
  float s = 0.f;
#pragma unroll
  for (int i = 0; i < 4; ++i) { lx[i] = x[tid + i * 256]; s += lx[i]; }
  red[tid] = s; __syncthreads();
  for (int off = 128; off > 0; off >>= 1) { if (tid < off) red[tid] += red[tid + off]; __syncthreads(); }
  const float mu = red[0] * (1.0f / DD);
  __syncthreads();
  s = 0.f;
#pragma unroll
  for (int i = 0; i < 4; ++i) { float d = lx[i] - mu; s += d * d; }
  red[tid] = s; __syncthreads();
  for (int off = 128; off > 0; off >>= 1) { if (tid < off) red[tid] += red[tid + off]; __syncthreads(); }
  const float rstd = rsqrtf(red[0] * (1.0f / DD) + EPSV);
#pragma unroll
  for (int i = 0; i < 4; ++i) {
    int c = tid + i * 256;
    y[c] = (lx[i] - mu) * rstd * w[c] + b[c];
  }
}

// ------------------------------------------------ WMMA bf16 GEMM, Y = act((X W^T + b)*scale) [+resid]
// X: [M,Kd] fp32 row-major, W: [N,Kd] fp32 row-major.
// One wave computes a 16x64 output tile: the A fragment is converted once per
// K-step and fed to 4 v_wmma_f32_16x16x32_bf16 (one per 16-col subtile).
// A-operand lane/K packing (contiguous, same order as B):
//   lanes 0-15  : K = 0..15; lanes 16-31 : K = 16..31
// B-operand: lane L holds column n=L%16, K = (L/16)*16 + 0..15.
// C/D: VGPR i, lane L -> row m = i + 8*(L/16), col n = L%16.
#define WMMA_NT(nn, ACC) {                                                          \
    const float4* wv = (const float4*)(wbase + (size_t)(nn) * 16 * Kd + k0);        \
    float4 u0 = wv[0], u1 = wv[1], u2 = wv[2], u3 = wv[3];                          \
    float wa[16] = { u0.x, u0.y, u0.z, u0.w, u1.x, u1.y, u1.z, u1.w,                \
                     u2.x, u2.y, u2.z, u2.w, u3.x, u3.y, u3.z, u3.w };              \
    v16bf bm;                                                                       \
    _Pragma("unroll") for (int j = 0; j < 16; ++j) bm[j] = (bf16_t)wa[j];           \
    ACC = __builtin_amdgcn_wmma_f32_16x16x32_bf16(false, a, false, bm,              \
                                                  (short)0, ACC, false, false); }

#define WMMA_EPI(nn, ACC) {                                                         \
    int n = ng * 64 + (nn) * 16 + l16;                                              \
    float bi = bias ? bias[n] : 0.0f;                                               \
    _Pragma("unroll") for (int i = 0; i < 8; ++i) {                                 \
      int m = mt * 16 + i + half * 8;                                               \
      float vv = (ACC[i] + bi) * scale;                                             \
      if (act == 1) vv = sigmoidf_(vv);                                             \
      if (resid) vv += resid[(size_t)m * N + n];                                    \
      Y[(size_t)m * N + n] = vv; } }

__global__ void gemm_bf16_wmma(const float* __restrict__ X, const float* __restrict__ W,
                               const float* __restrict__ bias, const float* __restrict__ resid,
                               float* __restrict__ Y, int M, int N, int Kd,
                               float scale, int act) {
  const int ngr  = N >> 6;                      // 64-column groups
  const int mt   = blockIdx.x / ngr;
  const int ng   = blockIdx.x % ngr;
  const int lane = threadIdx.x & 31;
  const int half = lane >> 4;
  const int l16  = lane & 15;

  const float* xrow  = X + (size_t)(mt * 16 + l16) * Kd;
  const float* wbase = W + (size_t)(ng * 64 + l16) * Kd + half * 16;

  v8f acc0 = {}, acc1 = {}, acc2 = {}, acc3 = {};
  for (int k0 = 0; k0 < Kd; k0 += 32) {
    // prefetch next K tile (global_prefetch_b8; speculative, drop if OOB)
    __builtin_prefetch(xrow + k0 + 32, 0, 1);
    __builtin_prefetch(wbase + k0 + 32, 0, 1);
    // A fragment: one 16-float contiguous chunk per lane (K = half*16 + 0..15), converted once
    const float4* xv0 = (const float4*)(xrow + k0 + half * 16);
    float4 t0 = xv0[0], t1 = xv0[1], t2 = xv0[2], t3 = xv0[3];
    float xa[16] = { t0.x, t0.y, t0.z, t0.w, t1.x, t1.y, t1.z, t1.w,
                     t2.x, t2.y, t2.z, t2.w, t3.x, t3.y, t3.z, t3.w };
    v16bf a;
#pragma unroll
    for (int j = 0; j < 16; ++j) a[j] = (bf16_t)xa[j];
    WMMA_NT(0, acc0)
    WMMA_NT(1, acc1)
    WMMA_NT(2, acc2)
    WMMA_NT(3, acc3)
  }
  WMMA_EPI(0, acc0)
  WMMA_EPI(1, acc1)
  WMMA_EPI(2, acc2)
  WMMA_EPI(3, acc3)
}

// ------------------------------------------------ causal conv (feature axis) + SiLU
__global__ void conv_silu_kernel(const float* __restrict__ xt, const float* __restrict__ cw,
                                 const float* __restrict__ cb, float* __restrict__ xc) {
  const int idx = blockIdx.x * blockDim.x + threadIdx.x;
  if (idx >= BB * PP) return;
  const int b = idx / PP, p = idx - b * PP;
  const float* xr = xt + (size_t)b * PP;
  float s = cb[0];
#pragma unroll
  for (int j = 0; j < KCONV; ++j) {
    int pp = p + j - (KCONV - 1);
    if (pp >= 0) s += cw[j] * xr[pp];
  }
  xc[idx] = siluf_(s);
}

// ------------------------------------------------ i/f gate dots + stabilized exp gating
__global__ void gates_kernel(const float* __restrict__ xc,
                             const float* __restrict__ Wi, const float* __restrict__ Wib,
                             const float* __restrict__ Wf, const float* __restrict__ Wfb,
                             const float* __restrict__ m_tm1,
                             float* __restrict__ ig, float* __restrict__ fg,
                             float* __restrict__ m_out) {
  __shared__ float ri[256], rf[256];
  const int bh = blockIdx.x;
  const int b = bh >> 3, h = bh & 7;
  const int tid = threadIdx.x;
  const float* x  = xc + (size_t)b * PP;
  const float* wi = Wi + (size_t)h * PP;
  const float* wf = Wf + (size_t)h * PP;
  float ai = 0.f, af = 0.f;
  for (int p = tid; p < PP; p += 256) {
    float xv = x[p];
    ai += xv * wi[p];
    af += xv * wf[p];
  }
  ri[tid] = ai; rf[tid] = af; __syncthreads();
  for (int off = 128; off > 0; off >>= 1) {
    if (tid < off) { ri[tid] += ri[tid + off]; rf[tid] += rf[tid + off]; }
    __syncthreads();
  }
  if (tid == 0) {
    float it = ri[0] + Wib[h];
    float ft = rf[0] + Wfb[h];
    float m0 = m_tm1[bh];
    float mt = fmaxf(ft + m0, it);
    m_out[bh] = mt;
    ig[bh] = __expf(it - mt);
    fg[bh] = __expf(ft + m0 - mt);
  }
}

// ------------------------------------------------ n_t update + denominator
__global__ void n_den_kernel(const float* __restrict__ n_tm1, const float* __restrict__ kk,
                             const float* __restrict__ q, const float* __restrict__ ig,
                             const float* __restrict__ fg,
                             float* __restrict__ n_out, float* __restrict__ den) {
  __shared__ float red[128];
  const int bh = blockIdx.x;
  const int tid = threadIdx.x;
  const size_t off = (size_t)bh * HDIM + tid;
  const float nv = fg[bh] * n_tm1[off] + ig[bh] * kk[off];
  n_out[off] = nv;
  red[tid] = nv * q[off]; __syncthreads();
  for (int o = 64; o > 0; o >>= 1) { if (tid < o) red[tid] += red[tid + o]; __syncthreads(); }
  if (tid == 0) den[bh] = fmaxf(red[0], 1.0f);
}

// ------------------------------------------------ cell update (HBM-bound) fused with num = c_t q
// One wave per (b,h,d) row of 128 fp32. The 268 MB of state traffic is streamed
// with non-temporal b128 load/store (TH=NT) so it never evicts L2-resident
// weights/activations; k/q stay regular (reused 128x per (b,h)).
__global__ void cell_kernel(const float* __restrict__ c_tm1, const float* __restrict__ v,
                            const float* __restrict__ kk, const float* __restrict__ q,
                            const float* __restrict__ ig, const float* __restrict__ fg,
                            float* __restrict__ c_out, float* __restrict__ num) {
  const int wave = threadIdx.x >> 5;
  const int lane = threadIdx.x & 31;
  const int row  = blockIdx.x * 8 + wave;   // [0, B*H*HD)
  const int bh   = row >> 7;
  const int d    = row & (HDIM - 1);
  const float f  = fg[bh], i = ig[bh];
  const float iv = i * v[(size_t)bh * HDIM + d];
  const size_t kqo = (size_t)bh * HDIM + lane * 4;
  const v4f kv = *(const v4f*)(kk + kqo);
  const v4f qv = *(const v4f*)(q + kqo);
  const size_t co = (size_t)row * HDIM + lane * 4;
  const v4f cv = __builtin_nontemporal_load((const v4f*)(c_tm1 + co));
  v4f cn;
  cn.x = f * cv.x + iv * kv.x;
  cn.y = f * cv.y + iv * kv.y;
  cn.z = f * cv.z + iv * kv.z;
  cn.w = f * cv.w + iv * kv.w;
  __builtin_nontemporal_store(cn, (v4f*)(c_out + co));
  float s = cn.x * qv.x + cn.y * qv.y + cn.z * qv.z + cn.w * qv.w;
#pragma unroll
  for (int o = 16; o > 0; o >>= 1) s += __shfl_xor(s, o, 32);
  if (lane == 0) num[row] = s;
}

// ------------------------------------------------ h_t + GroupNorm + skip + silu(r) gate
__global__ void gnorm_kernel(const float* __restrict__ num, const float* __restrict__ den,
                             const float* __restrict__ o, const float* __restrict__ gnw,
                             const float* __restrict__ gnb, const float* __restrict__ skip,
                             const float* __restrict__ r, float* __restrict__ g) {
  __shared__ float red[128];
  const int bh = blockIdx.x;
  const int j  = threadIdx.x;
  const int h  = bh & 7;
  const size_t off = (size_t)bh * HDIM + j;     // [B,H,HD] flat == [B,HID] flat
  const float ht = o[off] * (num[off] * __builtin_amdgcn_rcpf(den[bh]));
  red[j] = ht; __syncthreads();
  for (int s = 64; s > 0; s >>= 1) { if (j < s) red[j] += red[j + s]; __syncthreads(); }
  const float mu = red[0] * (1.0f / HDIM);
  __syncthreads();
  const float dlt = ht - mu;
  red[j] = dlt * dlt; __syncthreads();
  for (int s = 64; s > 0; s >>= 1) { if (j < s) red[j] += red[j + s]; __syncthreads(); }
  const float rstd = rsqrtf(red[0] * (1.0f / HDIM) + EPSV);
  const float hn = dlt * rstd * gnw[h * HDIM + j] + gnb[h * HDIM + j];
  g[off] = (hn + skip[off]) * siluf_(r[off]);
}

// ================================================================ launch
extern "C" void kernel_launch(void* const* d_in, const int* in_sizes, int n_in,
                              void* d_out, int out_size, void* d_ws, size_t ws_size,
                              hipStream_t stream) {
  const float* seq    = (const float*)d_in[0];
  const float* c_tm1  = (const float*)d_in[1];
  const float* n_tm1  = (const float*)d_in[2];
  const float* m_tm1  = (const float*)d_in[3];
  const float* ln_w   = (const float*)d_in[4];
  const float* ln_b   = (const float*)d_in[5];
  const float* gn_w   = (const float*)d_in[6];
  const float* gn_b   = (const float*)d_in[7];
  const float* up_l_w = (const float*)d_in[8];
  const float* up_l_b = (const float*)d_in[9];
  const float* up_r_w = (const float*)d_in[10];
  const float* up_r_b = (const float*)d_in[11];
  const float* down_w = (const float*)d_in[12];
  const float* down_b = (const float*)d_in[13];
  const float* Wi_w   = (const float*)d_in[14];
  const float* Wi_b   = (const float*)d_in[15];
  const float* Wf_w   = (const float*)d_in[16];
  const float* Wf_b   = (const float*)d_in[17];
  const float* Wo_w   = (const float*)d_in[18];
  const float* Wo_b   = (const float*)d_in[19];
  const float* Wq_w   = (const float*)d_in[20];
  const float* Wq_b   = (const float*)d_in[21];
  const float* Wk_w   = (const float*)d_in[22];
  const float* Wk_b   = (const float*)d_in[23];
  const float* Wv_w   = (const float*)d_in[24];
  const float* Wv_b   = (const float*)d_in[25];
  const float* conv_w = (const float*)d_in[26];
  const float* conv_b = (const float*)d_in[27];
  const float* skip_w = (const float*)d_in[28];

  // d_out = concat(out [B,D], c_t [B,H,HD,HD], n_t [B,H,HD], m_t [B,H])
  float* out_  = (float*)d_out;
  float* c_out = out_ + (size_t)BB * DD;
  float* n_out = c_out + (size_t)BB * HH * HDIM * HDIM;
  float* m_out = n_out + (size_t)BB * HH * HDIM;

  float* ws   = (float*)d_ws;
  float* x_n  = ws;
  float* x_t  = x_n  + (size_t)BB * DD;
  float* x_c  = x_t  + (size_t)BB * PP;
  float* r_t  = x_c  + (size_t)BB * PP;
  float* q    = r_t  + (size_t)BB * HIDD;
  float* kbuf = q    + (size_t)BB * HIDD;
  float* v    = kbuf + (size_t)BB * HIDD;
  float* o    = v    + (size_t)BB * HIDD;
  float* skip = o    + (size_t)BB * HIDD;
  float* num  = skip + (size_t)BB * HIDD;
  float* g    = num  + (size_t)BB * HIDD;
  float* ig   = g    + (size_t)BB * HIDD;
  float* fg   = ig   + (size_t)BB * HH;
  float* den  = fg   + (size_t)BB * HH;

  const float kscale = 0.08838834764831845f;  // 1/sqrt(128)

  ln_kernel<<<BB, 256, 0, stream>>>(seq, ln_w, ln_b, x_n);

  gemm_bf16_wmma<<<(BB / 16) * (PP / 64),   32, 0, stream>>>(x_n, up_l_w, up_l_b, nullptr, x_t, BB, PP,   DD, 1.0f, 0);
  gemm_bf16_wmma<<<(BB / 16) * (HIDD / 64), 32, 0, stream>>>(x_n, up_r_w, up_r_b, nullptr, r_t, BB, HIDD, DD, 1.0f, 0);

  conv_silu_kernel<<<(BB * PP) / 256, 256, 0, stream>>>(x_t, conv_w, conv_b, x_c);

  gemm_bf16_wmma<<<(BB / 16) * (HIDD / 64), 32, 0, stream>>>(x_c, Wq_w,  Wq_b,  nullptr, q,    BB, HIDD, PP, 1.0f,   0);
  gemm_bf16_wmma<<<(BB / 16) * (HIDD / 64), 32, 0, stream>>>(x_c, Wk_w,  Wk_b,  nullptr, kbuf, BB, HIDD, PP, kscale, 0);
  gemm_bf16_wmma<<<(BB / 16) * (HIDD / 64), 32, 0, stream>>>(x_t, Wv_w,  Wv_b,  nullptr, v,    BB, HIDD, PP, 1.0f,   0);
  gemm_bf16_wmma<<<(BB / 16) * (HIDD / 64), 32, 0, stream>>>(x_t, Wo_w,  Wo_b,  nullptr, o,    BB, HIDD, PP, 1.0f,   1);
  gemm_bf16_wmma<<<(BB / 16) * (HIDD / 64), 32, 0, stream>>>(x_c, skip_w, nullptr, nullptr, skip, BB, HIDD, PP, 1.0f, 0);

  gates_kernel<<<BB * HH, 256, 0, stream>>>(x_c, Wi_w, Wi_b, Wf_w, Wf_b, m_tm1, ig, fg, m_out);
  n_den_kernel<<<BB * HH, HDIM, 0, stream>>>(n_tm1, kbuf, q, ig, fg, n_out, den);
  cell_kernel<<<(BB * HH * HDIM) / 8, 256, 0, stream>>>(c_tm1, v, kbuf, q, ig, fg, c_out, num);
  gnorm_kernel<<<BB * HH, HDIM, 0, stream>>>(num, den, o, gn_w, gn_b, skip, r_t, g);

  gemm_bf16_wmma<<<(BB / 16) * (DD / 64), 32, 0, stream>>>(g, down_w, down_b, seq, out_, BB, DD, HIDD, 1.0f, 0);
}
